// ASGSCriterion_85057532330570
// MI455X (gfx1250) — compile-verified, matrix operand order checked by reading, not run on hardware
//
#include <hip/hip_runtime.h>
#include <math.h>

typedef __attribute__((ext_vector_type(16))) _Float16 v16h;
typedef __attribute__((ext_vector_type(8)))  _Float16 v8h;
typedef __attribute__((ext_vector_type(8)))  float    v8f;

#define BB  32
#define QQ  900
#define NM  300
#define DD  256
#define CC  90
#define NCC 91
#define KB  5
#define MK  5
#define CKK 450            // CC * KB
#define TAUF 0.1f
#define ALPHAF 0.25f
#define NEGINF (-1000000000.0f)
#define SIMBU_STRIDE 913   // odd stride -> conflict-free LDS row scans

// ---- WMMA helpers (CDNA5 16x16x32 f16 -> f32) -------------------------------
__device__ inline v8f wmma_f16(v16h a, v16h b, v8f c) {
  return __builtin_amdgcn_wmma_f32_16x16x32_f16(
      /*neg_a=*/false, a, /*neg_b=*/false, b,
      /*c_mod=*/(short)0, c, /*reuse_a=*/false, /*reuse_b=*/false);
}

// Load one 16-half fragment for lane (group g = lane>>4) from a row-major
// [16][K] tile row.  halves 0..7  -> K = k0 + g*8 + 0..7
//                    halves 8..15 -> K = k0 + 16 + g*8 + 0..7
__device__ inline v16h load_frag(const _Float16* rowbase, int k0, int g) {
  const v8h* p0 = (const v8h*)(rowbase + k0 + g * 8);
  const v8h* p1 = (const v8h*)(rowbase + k0 + 16 + g * 8);
  v8h lo = *p0, hi = *p1;
  v16h r;
#pragma unroll
  for (int i = 0; i < 8; ++i) { r[i] = lo[i]; r[i + 8] = hi[i]; }
  return r;
}

__device__ inline float wave_sum(float v) {
#pragma unroll
  for (int off = 16; off; off >>= 1) v += __shfl_xor(v, off, 32);
  return v;
}

// ---- prep: normalize prototypes (f32+f16), convert W_cls to f16 -------------
__global__ __launch_bounds__(32)
void k_prep_pw(const float* __restrict__ protos, const float* __restrict__ W,
               float* __restrict__ pn, _Float16* __restrict__ ph,
               _Float16* __restrict__ wh) {
  int blk = blockIdx.x, lane = threadIdx.x;
  if (blk < CC) {
    const float* row = protos + (size_t)blk * DD;
    float vals[8]; float ss = 0.f;
#pragma unroll
    for (int i = 0; i < 8; ++i) { float v = row[lane + 32 * i]; vals[i] = v; ss += v * v; }
    ss = wave_sum(ss);
    float inv = 1.0f / fmaxf(sqrtf(ss), 1e-12f);
#pragma unroll
    for (int i = 0; i < 8; ++i) {
      float v = vals[i] * inv;
      pn[(size_t)blk * DD + lane + 32 * i] = v;
      ph[(size_t)blk * DD + lane + 32 * i] = (_Float16)v;
    }
  } else {
    int r = blk - CC;
    const float* row = W + (size_t)r * DD;
#pragma unroll
    for (int i = 0; i < 8; ++i)
      wh[(size_t)r * DD + lane + 32 * i] = (_Float16)row[lane + 32 * i];
  }
}

// ---- prep: gather matched rows, normalize -> f16; mark is_m -----------------
__global__ __launch_bounds__(32)
void k_prep_matched(const float* __restrict__ obj, const int* __restrict__ msrc,
                    _Float16* __restrict__ mh, int* __restrict__ ism) {
  int t = blockIdx.x;                 // b*NM + n
  int b = t / NM;
  int src = msrc[t];
  const float* row = obj + ((size_t)b * QQ + src) * DD;
  int lane = threadIdx.x;
  float vals[8]; float ss = 0.f;
#pragma unroll
  for (int i = 0; i < 8; ++i) { float v = row[lane + 32 * i]; vals[i] = v; ss += v * v; }
  ss = wave_sum(ss);
  float inv = 1.0f / fmaxf(sqrtf(ss), 1e-12f);
#pragma unroll
  for (int i = 0; i < 8; ++i)
    mh[(size_t)t * DD + lane + 32 * i] = (_Float16)(vals[i] * inv);
  if (lane == 0) ism[b * QQ + src] = 1;
}

// ---- prep: normalize all obj rows -> f16 ------------------------------------
__global__ __launch_bounds__(32)
void k_prep_obj(const float* __restrict__ obj, _Float16* __restrict__ oh) {
  size_t t = blockIdx.x;              // b*QQ + q
  const float* row = obj + t * DD;
  int lane = threadIdx.x;
  float vals[8]; float ss = 0.f;
#pragma unroll
  for (int i = 0; i < 8; ++i) { float v = row[lane + 32 * i]; vals[i] = v; ss += v * v; }
  ss = wave_sum(ss);
  float inv = 1.0f / fmaxf(sqrtf(ss), 1e-12f);
#pragma unroll
  for (int i = 0; i < 8; ++i)
    oh[t * DD + lane + 32 * i] = (_Float16)(vals[i] * inv);
}

// ---- sims = matched_n @ protos^T  (WMMA, per-batch 300x90, K=256) -----------
__global__ __launch_bounds__(32)
void k_sims(const _Float16* __restrict__ mh, const _Float16* __restrict__ ph,
            float* __restrict__ sims) {
  int b = blockIdx.z, nt = blockIdx.y, ct = blockIdx.x;
  int lane = threadIdx.x, g = lane >> 4, idx = lane & 15;
  int arow = nt * 16 + idx; if (arow >= NM) arow = NM - 1;
  int bcol = ct * 16 + idx; if (bcol >= CC) bcol = CC - 1;
  const _Float16* ab = mh + ((size_t)b * NM + arow) * DD;
  const _Float16* bb = ph + (size_t)bcol * DD;
  v8f c = {0.f, 0.f, 0.f, 0.f, 0.f, 0.f, 0.f, 0.f};
#pragma unroll
  for (int ch = 0; ch < 8; ++ch) {
    v16h a = load_frag(ab, ch * 32, g);
    v16h bv = load_frag(bb, ch * 32, g);
    c = wmma_f16(a, bv, c);
  }
#pragma unroll
  for (int r = 0; r < 8; ++r) {
    int m = nt * 16 + r + 8 * g;
    int n = ct * 16 + idx;
    if (m < NM && n < CC) sims[((size_t)b * NM + m) * CC + n] = c[r];
  }
}

// ---- per-(b,c): top-KB by distance among rows with that label ---------------
__global__ void k_topk(const float* __restrict__ sims, const int* __restrict__ labels,
                       int* __restrict__ topi, int* __restrict__ cnt) {
  int t = blockIdx.x * blockDim.x + threadIdx.x;
  if (t >= BB * CC) return;
  int b = t / CC, c = t % CC;
  float bd[KB]; int bi[KB];
#pragma unroll
  for (int m = 0; m < KB; ++m) { bd[m] = -INFINITY; bi[m] = 0; }
  int cc = 0;
  for (int n = 0; n < NM; ++n) {
    if (labels[b * NM + n] != c) continue;
    ++cc;
    float d = 1.0f - sims[((size_t)b * NM + n) * CC + c];
    if (d > bd[KB - 1]) {
      int m = KB - 1;
      while (m > 0 && d > bd[m - 1]) { bd[m] = bd[m - 1]; bi[m] = bi[m - 1]; --m; }
      bd[m] = d; bi[m] = n;
    }
  }
  cnt[t] = cc;
#pragma unroll
  for (int m = 0; m < KB; ++m) topi[t * KB + m] = bi[m];
}

// ---- gather boundary rows (raw f32) + normalized f16 ------------------------
__global__ __launch_bounds__(32)
void k_boundary(const float* __restrict__ obj, const int* __restrict__ topi,
                const int* __restrict__ msrc, float* __restrict__ bnd,
                _Float16* __restrict__ bh) {
  int t = blockIdx.x;                 // b*CKK + ck
  int b = t / CKK, ck = t % CKK;
  int n = topi[(b * CC + ck / KB) * KB + (ck % KB)];
  int src = msrc[b * NM + n];
  const float* row = obj + ((size_t)b * QQ + src) * DD;
  int lane = threadIdx.x;
  float vals[8]; float ss = 0.f;
#pragma unroll
  for (int i = 0; i < 8; ++i) { float v = row[lane + 32 * i]; vals[i] = v; ss += v * v; }
  ss = wave_sum(ss);
  float inv = 1.0f / fmaxf(sqrtf(ss), 1e-12f);
#pragma unroll
  for (int i = 0; i < 8; ++i) {
    bnd[(size_t)t * DD + lane + 32 * i] = vals[i];
    bh[(size_t)t * DD + lane + 32 * i] = (_Float16)(vals[i] * inv);
  }
}

// ---- fused: sim_bu tile (16 rows x 900) via WMMA into LDS + top-MK ----------
__global__ __launch_bounds__(32)
void k_simbu(const _Float16* __restrict__ bh, const _Float16* __restrict__ oh,
             const int* __restrict__ ism, int* __restrict__ topq,
             int* __restrict__ topf) {
  __shared__ float sc[16 * SIMBU_STRIDE];
  int b = blockIdx.y, rt = blockIdx.x;
  int lane = threadIdx.x, g = lane >> 4, idx = lane & 15;
  int arow = rt * 16 + idx; if (arow >= CKK) arow = CKK - 1;
  const _Float16* ab = bh + ((size_t)b * CKK + arow) * DD;
  v16h afrag[8];
#pragma unroll
  for (int ch = 0; ch < 8; ++ch) afrag[ch] = load_frag(ab, ch * 32, g);

  for (int qt = 0; qt < 57; ++qt) {           // ceil(900/16)
    int bcol = qt * 16 + idx; int bcl = bcol >= QQ ? QQ - 1 : bcol;
    const _Float16* ob = oh + ((size_t)b * QQ + bcl) * DD;
    v8f c = {0.f, 0.f, 0.f, 0.f, 0.f, 0.f, 0.f, 0.f};
#pragma unroll
    for (int ch = 0; ch < 8; ++ch) {
      v16h bv = load_frag(ob, ch * 32, g);
      c = wmma_f16(afrag[ch], bv, c);
    }
    bool bad = (bcol >= QQ) || (ism[b * QQ + bcl] != 0 && bcol < QQ);
#pragma unroll
    for (int r = 0; r < 8; ++r) {
      float v = bad ? NEGINF : c[r];
      sc[(r + 8 * g) * SIMBU_STRIDE + qt * 16 + idx] = v;
    }
  }
  __syncthreads();

  if (lane < 16) {
    int row = rt * 16 + lane;
    if (row < CKK) {
      float bs[MK]; int bq[MK];
#pragma unroll
      for (int m = 0; m < MK; ++m) { bs[m] = -INFINITY; bq[m] = 0; }
      for (int q = 0; q < QQ; ++q) {
        float v = sc[lane * SIMBU_STRIDE + q];
        if (v > bs[MK - 1]) {
          int m = MK - 1;
          while (m > 0 && v > bs[m - 1]) { bs[m] = bs[m - 1]; bq[m] = bq[m - 1]; --m; }
          bs[m] = v; bq[m] = q;
        }
      }
#pragma unroll
      for (int m = 0; m < MK; ++m) {
        size_t o = ((size_t)b * CKK + row) * MK + m;
        topq[o] = bq[m];
        topf[o] = (bs[m] > 0.0f) ? 1 : 0;   // DELTA = 0
      }
    }
  }
}

// ---- g_bar = (boundary + sum w*un) / (1 + sum w); sg_valid ------------------
__global__ __launch_bounds__(32)
void k_gbar(const float* __restrict__ obj, const float* __restrict__ bnd,
            const int* __restrict__ topq, const int* __restrict__ topf,
            const int* __restrict__ cnt, _Float16* __restrict__ gh,
            int* __restrict__ sgv) {
  int t = blockIdx.x;                 // b*CKK + ck
  int b = t / CKK, ck = t % CKK;
  int c = ck / KB, k = ck % KB;
  bool bvalid = k < cnt[b * CC + c];
  int lane = threadIdx.x;
  float acc[8];
  const float* brow = bnd + (size_t)t * DD;
#pragma unroll
  for (int i = 0; i < 8; ++i) acc[i] = brow[lane + 32 * i];
  float wsum = 0.f; bool anyw = false;
#pragma unroll
  for (int m = 0; m < MK; ++m) {
    size_t o = (size_t)t * MK + m;
    if (bvalid && topf[o]) {
      anyw = true; wsum += 1.0f;
      const float* un = obj + ((size_t)b * QQ + topq[o]) * DD;
#pragma unroll
      for (int i = 0; i < 8; ++i) acc[i] += un[lane + 32 * i];
    }
  }
  float inv = 1.0f / (1.0f + wsum);
#pragma unroll
  for (int i = 0; i < 8; ++i)
    gh[(size_t)t * DD + lane + 32 * i] = (_Float16)(acc[i] * inv);
  if (lane == 0) sgv[t] = (anyw && bvalid) ? 1 : 0;
}

// ---- logits = g_bar @ W^T + b  (WMMA, 14400x91, K=256) ----------------------
__global__ __launch_bounds__(32)
void k_logits(const _Float16* __restrict__ gh, const _Float16* __restrict__ wh,
              const float* __restrict__ bcls, float* __restrict__ logits) {
  int rt = blockIdx.y, ct = blockIdx.x;
  int lane = threadIdx.x, g = lane >> 4, idx = lane & 15;
  int arow = rt * 16 + idx;           // always < 14400
  int bcol = ct * 16 + idx; if (bcol >= NCC) bcol = NCC - 1;
  const _Float16* ab = gh + (size_t)arow * DD;
  const _Float16* bb = wh + (size_t)bcol * DD;
  v8f acc = {0.f, 0.f, 0.f, 0.f, 0.f, 0.f, 0.f, 0.f};
#pragma unroll
  for (int ch = 0; ch < 8; ++ch) {
    v16h a = load_frag(ab, ch * 32, g);
    v16h bv = load_frag(bb, ch * 32, g);
    acc = wmma_f16(a, bv, acc);
  }
#pragma unroll
  for (int r = 0; r < 8; ++r) {
    int m = rt * 16 + r + 8 * g;
    int n = ct * 16 + idx;
    if (n < NCC) logits[(size_t)m * NCC + n] = acc[r] + bcls[n];
  }
}

// ---- focal loss per valid subgroup; atomic accumulate -----------------------
__global__ __launch_bounds__(32)
void k_focal(const float* __restrict__ logits, const int* __restrict__ sgv,
             float* __restrict__ acc_fl, int* __restrict__ acc_cnt) {
  int t = blockIdx.x;
  if (!sgv[t]) return;
  int lane = threadIdx.x;
  float s = 0.f;
  for (int j = lane; j < NCC; j += 32) {
    float x = logits[(size_t)t * NCC + j];
    float tt = (j == NCC - 1) ? 1.0f : 0.0f;
    float ls_x  = (x < 0.f) ? (x - log1pf(expf(x)))  : (-log1pf(expf(-x)));  // logsig(x)
    float ls_mx = (x > 0.f) ? (-x - log1pf(expf(-x))) : (-log1pf(expf(x)));  // logsig(-x)
    float ce = -(tt * ls_x + (1.0f - tt) * ls_mx);
    float p = 1.0f / (1.0f + expf(-x));
    float p_t = p * tt + (1.0f - p) * (1.0f - tt);
    float a_t = ALPHAF * tt + (1.0f - ALPHAF) * (1.0f - tt);
    float om = 1.0f - p_t;
    s += a_t * ce * om * om;                   // GAMMA = 2
  }
  s = wave_sum(s);
  if (lane == 0) { atomicAdd(acc_fl, s / (float)NCC); atomicAdd(acc_cnt, 1); }
}

// ---- per-column logsumexp over masked P (90 rows) and masked S (9600 rows) --
__global__ __launch_bounds__(256)
void k_lse(const float* __restrict__ pn, const float* __restrict__ sims,
           const int* __restrict__ labels, float* __restrict__ lse_neg) {
  int c = blockIdx.x, t = threadIdx.x;
  float m = -INFINITY, s = 0.0f;
  for (int r = t; r < CC; r += 256) {          // P column (exclude diag)
    if (r == c) continue;
    const float* pr = pn + (size_t)r * DD;
    const float* pc = pn + (size_t)c * DD;
    float d = 0.f;
    for (int i = 0; i < DD; ++i) d += pr[i] * pc[i];
    float v = d / TAUF;
    if (v > m) { s = s * expf(m - v) + 1.0f; m = v; } else s += expf(v - m);
  }
  for (int i = t; i < BB * NM; i += 256) {     // S column (exclude lab==c rows)
    if (labels[i] == c) continue;
    float v = sims[(size_t)i * CC + c] / TAUF;
    if (v > m) { s = s * expf(m - v) + 1.0f; m = v; } else s += expf(v - m);
  }
  __shared__ float sm[256], ssum[256];
  sm[t] = m; ssum[t] = s; __syncthreads();
  for (int off = 128; off; off >>= 1) {
    if (t < off) {
      float m2 = sm[t + off], s2 = ssum[t + off];
      float M = fmaxf(sm[t], m2);
      if (M > -INFINITY) {
        ssum[t] = ssum[t] * expf(sm[t] - M) + s2 * expf(m2 - M);
        sm[t] = M;
      }
    }
    __syncthreads();
  }
  if (t == 0) lse_neg[c] = sm[0] + logf(ssum[0]);
}

// ---- contrastive term per matched row; atomic accumulate --------------------
__global__ __launch_bounds__(256)
void k_cec(const float* __restrict__ sims, const int* __restrict__ labels,
           const float* __restrict__ lse_neg, float* __restrict__ acc_cec) {
  int i = blockIdx.x * 256 + threadIdx.x;
  float val = 0.f;
  if (i < BB * NM) {
    int lab = labels[i];
    float pos = sims[(size_t)i * CC + lab] / TAUF;
    float l = lse_neg[lab];
    float M = fmaxf(pos, l);
    float la = M + logf(expf(pos - M) + expf(l - M));
    val = -pos + la;
  }
  __shared__ float sh[256];
  sh[threadIdx.x] = val; __syncthreads();
  for (int off = 128; off; off >>= 1) {
    if (threadIdx.x < off) sh[threadIdx.x] += sh[threadIdx.x + off];
    __syncthreads();
  }
  if (threadIdx.x == 0) atomicAdd(acc_cec, sh[0]);
}

__global__ void k_final(const float* __restrict__ acc_fl, const int* __restrict__ acc_cnt,
                        const float* __restrict__ acc_cec, float* __restrict__ out) {
  int n = *acc_cnt; if (n < 1) n = 1;
  out[0] = (*acc_fl) / (float)n;
  out[1] = (*acc_cec) / (float)(BB * NM);
}

// =============================================================================
extern "C" void kernel_launch(void* const* d_in, const int* in_sizes, int n_in,
                              void* d_out, int out_size, void* d_ws, size_t ws_size,
                              hipStream_t stream) {
  (void)in_sizes; (void)n_in; (void)out_size; (void)ws_size;
  const float* obj    = (const float*)d_in[0];
  const float* protos = (const float*)d_in[1];
  const float* W      = (const float*)d_in[2];
  const float* bcls   = (const float*)d_in[3];
  const int*   msrc   = (const int*)d_in[4];
  const int*   mlab   = (const int*)d_in[5];
  float* out = (float*)d_out;

  char* base = (char*)d_ws;
  size_t off = 0;
  auto take = [&](size_t bytes) -> char* {
    char* p = base + off;
    off = (off + bytes + 255) & ~(size_t)255;
    return p;
  };
  _Float16* mh   = (_Float16*)take((size_t)BB * NM * DD * 2);
  _Float16* oh   = (_Float16*)take((size_t)BB * QQ * DD * 2);
  float*    pn   = (float*)   take((size_t)CC * DD * 4);
  _Float16* ph   = (_Float16*)take((size_t)CC * DD * 2);
  _Float16* wh   = (_Float16*)take((size_t)NCC * DD * 2);
  float*    sims = (float*)   take((size_t)BB * NM * CC * 4);
  int*      cnt  = (int*)     take((size_t)BB * CC * 4);
  int*      topi = (int*)     take((size_t)BB * CC * KB * 4);
  int*      ism  = (int*)     take((size_t)BB * QQ * 4);
  float*    bnd  = (float*)   take((size_t)BB * CKK * DD * 4);
  _Float16* bh   = (_Float16*)take((size_t)BB * CKK * DD * 2);
  int*      topq = (int*)     take((size_t)BB * CKK * MK * 4);
  int*      topf = (int*)     take((size_t)BB * CKK * MK * 4);
  _Float16* gh   = (_Float16*)take((size_t)BB * CKK * DD * 2);
  int*      sgv  = (int*)     take((size_t)BB * CKK * 4);
  float*    lg   = (float*)   take((size_t)BB * CKK * NCC * 4);
  float*    lse  = (float*)   take((size_t)CC * 4);
  float*    accf = (float*)   take(64);        // [0]=sum_fl, [1]=sum_cec
  int*      acci = (int*)accf + 2;             // count of sg_valid

  hipMemsetAsync(ism, 0, (size_t)BB * QQ * 4, stream);
  hipMemsetAsync(accf, 0, 64, stream);

  k_prep_pw     <<<CC + NCC, 32, 0, stream>>>(protos, W, pn, ph, wh);
  k_prep_matched<<<BB * NM, 32, 0, stream>>>(obj, msrc, mh, ism);
  k_prep_obj    <<<BB * QQ, 32, 0, stream>>>(obj, oh);
  k_sims        <<<dim3(6, 19, BB), 32, 0, stream>>>(mh, ph, sims);
  k_topk        <<<(BB * CC + 255) / 256, 256, 0, stream>>>(sims, mlab, topi, cnt);
  k_boundary    <<<BB * CKK, 32, 0, stream>>>(obj, topi, msrc, bnd, bh);
  k_simbu       <<<dim3(29, BB), 32, 0, stream>>>(bh, oh, ism, topq, topf);
  k_gbar        <<<BB * CKK, 32, 0, stream>>>(obj, bnd, topq, topf, cnt, gh, sgv);
  k_logits      <<<dim3(6, 900), 32, 0, stream>>>(gh, wh, bcls, lg);
  k_focal       <<<BB * CKK, 32, 0, stream>>>(lg, sgv, &accf[0], acci);
  k_lse         <<<CC, 256, 0, stream>>>(pn, sims, mlab, lse);
  k_cec         <<<(BB * NM + 255) / 256, 256, 0, stream>>>(sims, mlab, lse, &accf[1]);
  k_final       <<<1, 1, 0, stream>>>(&accf[0], acci, &accf[1], out);
}
